// CrossModalFusion_42726334661225
// MI455X (gfx1250) — matrix operand order repaired and checked by reading, not compiled, not run on hardware
//
#include <hip/hip_runtime.h>
#include <hip/hip_bf16.h>
#include <math.h>

// ---------------------------------------------------------------------------
// Problem constants (from reference)
// ---------------------------------------------------------------------------
constexpr int kD     = 1024;
constexpr int kDFF   = 4096;
constexpr int kH     = 16;
constexpr int kHD    = 64;
constexpr int kB     = 8;
constexpr int kLQ    = 512;
constexpr int kLK    = 1024;
constexpr int kDepth = 4;
constexpr int kMQ    = kB * kLQ;   // 4096 query rows
constexpr int kMKV   = kB * kLK;   // 8192 key/value rows

typedef __attribute__((ext_vector_type(16))) __bf16 v16bf;
typedef __attribute__((ext_vector_type(8)))  float  v8f;
typedef unsigned int v4u __attribute__((ext_vector_type(4)));
typedef int          v8i __attribute__((ext_vector_type(8)));
typedef int          v4i __attribute__((ext_vector_type(4)));

#if __has_builtin(__builtin_amdgcn_tensor_load_to_lds) && \
    __has_builtin(__builtin_amdgcn_s_wait_tensorcnt)
#define USE_TDM 1
#else
#define USE_TDM 0
#endif

struct U16x16 { uint4 lo, hi; };

__device__ __forceinline__ unsigned short f2bf(float f) {
    unsigned int u = __builtin_bit_cast(unsigned int, f);
    u += 0x7FFFu + ((u >> 16) & 1u);          // round-to-nearest-even
    return (unsigned short)(u >> 16);
}

__device__ __forceinline__ v8f v8f_zero() {
    v8f z;
#pragma unroll
    for (int i = 0; i < 8; ++i) z[i] = 0.0f;
    return z;
}

// D = A(16x32 bf16) x B(32x16 bf16) + C, fp32 accum
__device__ __forceinline__ v8f wmma_bf16(v16bf a, v16bf b, v8f c) {
    return __builtin_amdgcn_wmma_f32_16x16x32_bf16(
        /*neg_a=*/false, a, /*neg_b=*/false, b,
        /*c_mod=*/(short)0, c, /*reuse_a=*/false, /*reuse_b=*/false);
}

// ---------------------------------------------------------------------------
// Fragment loaders (ISA 7.12.2 layouts), tiles row-major [row][k] in LDS.
// ---------------------------------------------------------------------------
__device__ __forceinline__ v16bf load_frag_A(const unsigned short* lds, int stride,
                                             int row0, int k0, int lane) {
    int row = row0 + (lane & 15);
    int kb  = k0 + ((lane >> 4) << 3);
    const unsigned short* p = lds + row * stride + kb;
    U16x16 u;
    u.lo = *(const uint4*)(p);        // K = kb .. kb+7
    u.hi = *(const uint4*)(p + 16);   // K = kb+16 .. kb+23
    return __builtin_bit_cast(v16bf, u);
}

__device__ __forceinline__ v16bf load_frag_B(const unsigned short* lds, int stride,
                                             int n0, int k0, int lane) {
    int n  = n0 + (lane & 15);
    int kb = k0 + ((lane >> 4) << 3);
    const unsigned short* p = lds + n * stride + kb;
    U16x16 u;
    u.lo = *(const uint4*)(p);        // K = kb .. kb+7
    u.hi = *(const uint4*)(p + 16);   // K = kb+16 .. kb+23
    return __builtin_bit_cast(v16bf, u);
}

// ---------------------------------------------------------------------------
// fp32 -> bf16 elementwise convert
// ---------------------------------------------------------------------------
__global__ __launch_bounds__(256)
void convert_bf16_kernel(const float* __restrict__ in, unsigned short* __restrict__ out, int n) {
    int i = blockIdx.x * 256 + threadIdx.x;
    if (i < n) out[i] = f2bf(in[i]);
}

// ---------------------------------------------------------------------------
// LayerNorm over last dim (D=1024), output bf16.  One block (256 thr) per row.
// ---------------------------------------------------------------------------
__global__ __launch_bounds__(256)
void ln_kernel(const float* __restrict__ x, const float* __restrict__ g,
               const float* __restrict__ bb, unsigned short* __restrict__ out) {
    __shared__ float red[2][8];
    const int row = blockIdx.x;
    const int tid = threadIdx.x;
    const float* xr = x + (size_t)row * kD;
    float v[4];
    float s = 0.0f, ss = 0.0f;
#pragma unroll
    for (int j = 0; j < 4; ++j) {
        v[j] = xr[tid + j * 256];
        s  += v[j];
        ss += v[j] * v[j];
    }
#pragma unroll
    for (int msk = 16; msk > 0; msk >>= 1) {
        s  += __shfl_xor(s,  msk, 32);
        ss += __shfl_xor(ss, msk, 32);
    }
    if ((tid & 31) == 0) { red[0][tid >> 5] = s; red[1][tid >> 5] = ss; }
    __syncthreads();
    float ts = 0.0f, tss = 0.0f;
#pragma unroll
    for (int w = 0; w < 8; ++w) { ts += red[0][w]; tss += red[1][w]; }
    const float mean = ts * (1.0f / kD);
    const float var  = tss * (1.0f / kD) - mean * mean;
    const float rs   = rsqrtf(var + 1e-5f);
#pragma unroll
    for (int j = 0; j < 4; ++j) {
        const int c = tid + j * 256;
        out[(size_t)row * kD + c] = f2bf((v[j] - mean) * rs * g[c] + bb[c]);
    }
}

// ---------------------------------------------------------------------------
// GEMM tile staging.  LDS row stride = 40 halves (32 data + 8 pad).
// ---------------------------------------------------------------------------
constexpr int kLDT = 40;

// Flat pointer -> LDS byte offset (ISA 10.2: LDS aperture address = addr[31:0]).
__device__ __forceinline__ unsigned lds_byte_offset(const void* p) {
    return (unsigned)(unsigned long long)p;
}

// TDM: DMA a 128x32 bf16 tile (global row stride K elements) into LDS with the
// padded 40-half row layout (pad_interval: 16 DWORDs, pad_amount: 4 DWORDs).
__device__ __forceinline__ void tdm_load_tile(const unsigned short* gsrc,
                                              unsigned lds_off, int K) {
#if USE_TDM
    unsigned long long ga = (unsigned long long)gsrc;
    v4u g0;
    g0[0] = 1u;                                                 // count=1, user D#
    g0[1] = lds_off;                                            // lds_addr (bytes)
    g0[2] = (unsigned)(ga & 0xFFFFFFFFu);                       // global_addr[31:0]
    g0[3] = (unsigned)((ga >> 32) & 0x01FFFFFFu) | 0x80000000u; // addr[56:32] | type=2
    v8i g1;
    g1[0] = (int)((1u << 16)        // data_size = 2 bytes
                | (1u << 20)        // pad_enable
                | (3u << 22)        // pad_interval = 16 DWORDs (32 halves)
                | (3u << 25));      // pad_amount   = 4 DWORDs  (8 halves)
    g1[1] = (int)(32u  << 16);      // tensor_dim0 = 32
    g1[2] = (int)(128u << 16);      // tensor_dim1 = 128
    g1[3] = (int)(32u  << 16);      // tile_dim0   = 32
    g1[4] = 128;                    // tile_dim1   = 128 (tile_dim2 = 0)
    g1[5] = K;                      // tensor_dim0_stride = K elements
    g1[6] = 0;
    g1[7] = 0;
    v4i gz4 = {0, 0, 0, 0};
    v8i gz8 = {0, 0, 0, 0, 0, 0, 0, 0};
    __builtin_amdgcn_tensor_load_to_lds(g0, g1, gz4, gz4, gz8, 0);
#else
    (void)gsrc; (void)lds_off; (void)K;
#endif
}

// Fallback / bf16 A tile copy via VGPRs (256 threads, 128x32 tile).
__device__ __forceinline__ void load_tile_A(const unsigned short* __restrict__ A,
                                            int K, int m0, int kt,
                                            unsigned short* As, int tid) {
    const int row = tid >> 1;       // 0..127
    const int hf  = tid & 1;        // 0..1
    const uint4* src = (const uint4*)(A + (size_t)(m0 + row) * K + kt * 32 + hf * 16);
    uint4* dst = (uint4*)(&As[row * kLDT + hf * 16]);
    dst[0] = src[0];
    dst[1] = src[1];
}

// W tile: fp32 -> bf16 convert during LDS fill (256 threads, 128x32 tile).
__device__ __forceinline__ void load_tile_W(const float* __restrict__ W,
                                            int K, int n0, int kt,
                                            unsigned short* Ws, int tid) {
    const int row = tid >> 1;
    const int hf  = tid & 1;
    const float4* src = (const float4*)(W + (size_t)(n0 + row) * K + kt * 32 + hf * 16);
    unsigned short* dst = &Ws[row * kLDT + hf * 16];
#pragma unroll
    for (int j = 0; j < 4; ++j) {
        float4 f = src[j];
        unsigned int p0 = (unsigned int)f2bf(f.x) | ((unsigned int)f2bf(f.y) << 16);
        unsigned int p1 = (unsigned int)f2bf(f.z) | ((unsigned int)f2bf(f.w) << 16);
        *(unsigned int*)(dst + j * 4)     = p0;
        *(unsigned int*)(dst + j * 4 + 2) = p1;
    }
}

// ---------------------------------------------------------------------------
// GEMM:  O[M][N] = act( A[M][K](bf16) @ W[N][K]^T(fp32) + bias[N] )
// ACT: 0 = none, 1 = exact GELU.   OUT: 0 = bf16 store, 1 = fp32 += (residual)
// Block: 256 threads (8 waves).  Block tile 128x128, BK=32, double-buffered.
// A tile staged by the Tensor Data Mover (wave 0 issues, TENSORcnt-tracked);
// W tile converted fp32->bf16 through VGPRs by all waves in parallel.
// ---------------------------------------------------------------------------
template <int ACT, int OUT>
__global__ __launch_bounds__(256)
void gemm_bf16_kernel(const unsigned short* __restrict__ A,
                      const float* __restrict__ W,
                      const float* __restrict__ bias,
                      unsigned short* __restrict__ Obf,
                      float* __restrict__ Of32,
                      int M, int N, int K) {
    __shared__ unsigned short As[2][128 * kLDT];
    __shared__ unsigned short Ws[2][128 * kLDT];

    const int tid  = threadIdx.x;
    const int lane = tid & 31;
    const int wid  = tid >> 5;          // 0..7
    const int mw   = wid & 3;           // 0..3  -> 32-row strip
    const int nw   = wid >> 2;          // 0..1  -> 64-col strip
    const int n0   = blockIdx.x * 128;
    const int m0   = blockIdx.y * 128;

    v8f acc[2][4];
#pragma unroll
    for (int i = 0; i < 2; ++i)
#pragma unroll
        for (int t = 0; t < 4; ++t) acc[i][t] = v8f_zero();

    const int ksteps = K >> 5;

    // prologue: stage K-tile 0
#if USE_TDM
    if (wid == 0)
        tdm_load_tile(A + (size_t)m0 * K, lds_byte_offset(&As[0][0]), K);
    load_tile_W(W, K, n0, 0, Ws[0], tid);
    if (wid == 0) __builtin_amdgcn_s_wait_tensorcnt(0);
#else
    load_tile_A(A, K, m0, 0, As[0], tid);
    load_tile_W(W, K, n0, 0, Ws[0], tid);
#endif
    __syncthreads();

    int buf = 0;
    for (int kt = 0; kt < ksteps; ++kt) {
        // stage next K-tile into the other buffer while computing this one
        if (kt + 1 < ksteps) {
#if USE_TDM
            if (wid == 0)
                tdm_load_tile(A + (size_t)m0 * K + (kt + 1) * 32,
                              lds_byte_offset(&As[buf ^ 1][0]), K);
#else
            load_tile_A(A, K, m0, kt + 1, As[buf ^ 1], tid);
#endif
            load_tile_W(W, K, n0, kt + 1, Ws[buf ^ 1], tid);
        }
        // prefetch K+2 tile of W toward L2/WGP$ (global_prefetch_b8)
        if (kt + 2 < ksteps) {
            const int row = tid >> 1, hf = tid & 1;
            __builtin_prefetch(W + (size_t)(n0 + row) * K + (kt + 2) * 32 + hf * 16, 0, 0);
        }

        v16bf af0 = load_frag_A(As[buf], kLDT, mw * 32,      0, lane);
        v16bf af1 = load_frag_A(As[buf], kLDT, mw * 32 + 16, 0, lane);
#pragma unroll
        for (int t = 0; t < 4; ++t) {
            v16bf bf = load_frag_B(Ws[buf], kLDT, nw * 64 + t * 16, 0, lane);
            acc[0][t] = wmma_bf16(af0, bf, acc[0][t]);
            acc[1][t] = wmma_bf16(af1, bf, acc[1][t]);
        }

#if USE_TDM
        if (wid == 0 && kt + 1 < ksteps) __builtin_amdgcn_s_wait_tensorcnt(0);
#endif
        __syncthreads();
        buf ^= 1;
    }

    // --- epilogue ---
    const int colh  = lane & 15;
    const int rbase = (lane >> 4) << 3;         // 0 or 8
#pragma unroll
    for (int i = 0; i < 2; ++i) {
#pragma unroll
        for (int t = 0; t < 4; ++t) {
            const int gn = n0 + nw * 64 + t * 16 + colh;
            const float bv = bias[gn];
#pragma unroll
            for (int r = 0; r < 8; ++r) {
                const int gm = m0 + mw * 32 + i * 16 + rbase + r;
                float v = acc[i][t][r] + bv;
                if (ACT == 1) v = 0.5f * v * (1.0f + erff(v * 0.70710678118654752f));
                const size_t idx = (size_t)gm * N + gn;
                if (OUT == 0) Obf[idx] = f2bf(v);
                else          Of32[idx] += v;
            }
        }
    }
}

// ---------------------------------------------------------------------------
// Flash attention:  one block = (b, h, 64 query rows).  4 waves x 16 q-rows.
// K-chunks of 64 keys; online softmax; both matmuls via wmma bf16.
// ---------------------------------------------------------------------------
__global__ __launch_bounds__(128)
void attn_kernel(const unsigned short* __restrict__ Q,   // [B*Lq][D] bf16
                 const unsigned short* __restrict__ Kb,  // [B*Lk][D] bf16
                 const unsigned short* __restrict__ Vb,  // [B*Lk][D] bf16
                 unsigned short* __restrict__ ctx) {     // [B*Lq][D] bf16
    constexpr int LDH = 72;                    // 64 + 8 pad
    __shared__ unsigned short Qs[64 * LDH];
    __shared__ unsigned short Ks[64 * LDH];
    __shared__ unsigned short Vt[64 * LDH];    // transposed: [hd][key]
    __shared__ unsigned short Ps[4][16 * LDH]; // per-wave probabilities

    const int qt = blockIdx.x;                 // 0..7
    const int h  = blockIdx.y;                 // 0..15
    const int bb = blockIdx.z;                 // 0..7
    const int tid  = threadIdx.x;
    const int lane = tid & 31;
    const int wid  = tid >> 5;
    const float scale = 0.125f;                // 1/sqrt(64)

    const size_t qbase  = ((size_t)bb * kLQ + qt * 64) * kD + h * kHD;
    const size_t kvbase = ((size_t)bb * kLK) * kD + h * kHD;

    const int lrow = tid >> 1;                 // 0..63
    const int lhf  = tid & 1;                  // 0..1 (32-element half of hd)

    // load Q tile (64 x 64)
    {
        const uint4* src = (const uint4*)(Q + qbase + (size_t)lrow * kD + lhf * 32);
        uint4* dst = (uint4*)(&Qs[lrow * LDH + lhf * 32]);
#pragma unroll
        for (int j = 0; j < 4; ++j) dst[j] = src[j];
    }
    __syncthreads();

    // Q fragments are invariant across the key loop
    const v16bf qa0 = load_frag_A(Qs, LDH, wid * 16, 0, lane);
    const v16bf qa1 = load_frag_A(Qs, LDH, wid * 16, 32, lane);

    float mrow[8], lsum[8];
#pragma unroll
    for (int r = 0; r < 8; ++r) { mrow[r] = -1e30f; lsum[r] = 0.0f; }
    v8f acc[4];
#pragma unroll
    for (int t = 0; t < 4; ++t) acc[t] = v8f_zero();

    for (int kc = 0; kc < kLK / 64; ++kc) {
        __syncthreads();
        // K chunk (64 keys x 64 hd), row-major
        {
            const uint4* src = (const uint4*)(Kb + kvbase + (size_t)(kc * 64 + lrow) * kD + lhf * 32);
            uint4* dst = (uint4*)(&Ks[lrow * LDH + lhf * 32]);
#pragma unroll
            for (int j = 0; j < 4; ++j) dst[j] = src[j];
        }
        // V chunk, transposed into Vt[hd][key]
        {
            const uint4* src = (const uint4*)(Vb + kvbase + (size_t)(kc * 64 + lrow) * kD + lhf * 32);
#pragma unroll
            for (int j = 0; j < 4; ++j) {
                uint4 q = src[j];
                unsigned int w[4] = {q.x, q.y, q.z, q.w};
#pragma unroll
                for (int e = 0; e < 4; ++e) {
                    const int d0 = lhf * 32 + j * 8 + e * 2;
                    Vt[(d0    ) * LDH + lrow] = (unsigned short)(w[e] & 0xFFFFu);
                    Vt[(d0 + 1) * LDH + lrow] = (unsigned short)(w[e] >> 16);
                }
            }
        }
        __syncthreads();

        // S = Q @ K^T  (16 q-rows x 64 keys per wave)
        v8f s[4];
#pragma unroll
        for (int t = 0; t < 4; ++t) {
            s[t] = v8f_zero();
            v16bf kb0 = load_frag_B(Ks, LDH, t * 16, 0, lane);
            v16bf kb1 = load_frag_B(Ks, LDH, t * 16, 32, lane);
            s[t] = wmma_bf16(qa0, kb0, s[t]);
            s[t] = wmma_bf16(qa1, kb1, s[t]);
        }

        // online softmax (row r lives in element r across 16 lanes of a half)
#pragma unroll
        for (int r = 0; r < 8; ++r) {
            float x0 = fmaxf(fmaxf(s[0][r], s[1][r]), fmaxf(s[2][r], s[3][r])) * scale;
#pragma unroll
            for (int msk = 8; msk > 0; msk >>= 1) x0 = fmaxf(x0, __shfl_xor(x0, msk, 32));
            const float nm = fmaxf(mrow[r], x0);
            float rsum = 0.0f;
#pragma unroll
            for (int t = 0; t < 4; ++t) {
                float pv = __expf(s[t][r] * scale - nm);
                s[t][r] = pv;
                rsum += pv;
            }
#pragma unroll
            for (int msk = 8; msk > 0; msk >>= 1) rsum += __shfl_xor(rsum, msk, 32);
            const float f = __expf(mrow[r] - nm);
            lsum[r] = lsum[r] * f + rsum;
            mrow[r] = nm;
#pragma unroll
            for (int t = 0; t < 4; ++t) acc[t][r] *= f;
        }

        // stage P (D-layout -> row-major LDS) for the A-fragment of P@V
        {
            unsigned short* pw = &Ps[wid][0];
            const int prow = (lane >> 4) << 3;
            const int pc   = lane & 15;
#pragma unroll
            for (int t = 0; t < 4; ++t)
#pragma unroll
                for (int r = 0; r < 8; ++r)
                    pw[(prow + r) * LDH + t * 16 + pc] = f2bf(s[t][r]);
        }

        // ctx += P @ V
        v16bf pa0 = load_frag_A(&Ps[wid][0], LDH, 0, 0, lane);
        v16bf pa1 = load_frag_A(&Ps[wid][0], LDH, 0, 32, lane);
#pragma unroll
        for (int t = 0; t < 4; ++t) {
            v16bf vb0 = load_frag_B(Vt, LDH, t * 16, 0, lane);
            v16bf vb1 = load_frag_B(Vt, LDH, t * 16, 32, lane);
            acc[t] = wmma_bf16(pa0, vb0, acc[t]);
            acc[t] = wmma_bf16(pa1, vb1, acc[t]);
        }
    }

    // write ctx (normalize by l)
    const int colh  = lane & 15;
    const int rbase = (lane >> 4) << 3;
#pragma unroll
    for (int t = 0; t < 4; ++t) {
#pragma unroll
        for (int r = 0; r < 8; ++r) {
            const int gq = bb * kLQ + qt * 64 + wid * 16 + rbase + r;
            const int gc = h * kHD + t * 16 + colh;
            ctx[(size_t)gq * kD + gc] = f2bf(acc[t][r] / lsum[r]);
        }
    }
}

// ---------------------------------------------------------------------------
// Host-side orchestration
// ---------------------------------------------------------------------------
extern "C" void kernel_launch(void* const* d_in, const int* in_sizes, int n_in,
                              void* d_out, int out_size, void* d_ws, size_t ws_size,
                              hipStream_t stream) {
    const float* txt       = (const float*)d_in[0];
    const float* img       = (const float*)d_in[1];
    const float* in_proj_w = (const float*)d_in[2];
    const float* in_proj_b = (const float*)d_in[3];
    const float* out_w     = (const float*)d_in[4];
    const float* out_b     = (const float*)d_in[5];
    const float* ln1_g     = (const float*)d_in[6];
    const float* ln1_b     = (const float*)d_in[7];
    const float* ln2_g     = (const float*)d_in[8];
    const float* ln2_b     = (const float*)d_in[9];
    const float* ff1_w     = (const float*)d_in[10];
    const float* ff1_b     = (const float*)d_in[11];
    const float* ff2_w     = (const float*)d_in[12];
    const float* ff2_b     = (const float*)d_in[13];

    float* x = (float*)d_out;                  // residual stream (fp32)

    // workspace layout (bf16 halves)
    unsigned short* ws     = (unsigned short*)d_ws;
    unsigned short* img_bf = ws;                                    // 8M
    unsigned short* qin_bf = img_bf + (size_t)kMKV * kD;            // 4M (also LN2 out)
    unsigned short* Qbf    = qin_bf + (size_t)kMQ * kD;             // 4M
    unsigned short* Kbf    = Qbf    + (size_t)kMQ * kD;             // 8M
    unsigned short* Vbf    = Kbf    + (size_t)kMKV * kD;            // 8M
    unsigned short* ctxbf  = Vbf    + (size_t)kMKV * kD;            // 4M
    unsigned short* ffbf   = Kbf;   // aliases K+V (16M) — dead by FF time

    // x = txt (residual init)
    (void)hipMemcpyAsync(x, txt, (size_t)kMQ * kD * sizeof(float),
                         hipMemcpyDeviceToDevice, stream);

    // img -> bf16 once
    {
        const int n = kMKV * kD;
        convert_bf16_kernel<<<n / 256, 256, 0, stream>>>(img, img_bf, n);
    }

    for (int l = 0; l < kDepth; ++l) {
        const float* wq = in_proj_w + (size_t)l * 3 * kD * kD;
        const float* wk = wq + (size_t)kD * kD;
        const float* wv = wk + (size_t)kD * kD;
        const float* bq = in_proj_b + (size_t)l * 3 * kD;
        const float* bk = bq + kD;
        const float* bv = bk + kD;
        const float* ow = out_w + (size_t)l * kD * kD;
        const float* ob = out_b + (size_t)l * kD;
        const float* w1 = ff1_w + (size_t)l * kDFF * kD;
        const float* b1 = ff1_b + (size_t)l * kDFF;
        const float* w2 = ff2_w + (size_t)l * kD * kDFF;
        const float* b2 = ff2_b + (size_t)l * kD;

        // q_in = LN1(x)  (bf16)
        ln_kernel<<<kMQ, 256, 0, stream>>>(x, ln1_g + (size_t)l * kD,
                                           ln1_b + (size_t)l * kD, qin_bf);

        // Q / K / V projections
        gemm_bf16_kernel<0, 0><<<dim3(kD / 128, kMQ / 128), 256, 0, stream>>>(
            qin_bf, wq, bq, Qbf, nullptr, kMQ, kD, kD);
        gemm_bf16_kernel<0, 0><<<dim3(kD / 128, kMKV / 128), 256, 0, stream>>>(
            img_bf, wk, bk, Kbf, nullptr, kMKV, kD, kD);
        gemm_bf16_kernel<0, 0><<<dim3(kD / 128, kMKV / 128), 256, 0, stream>>>(
            img_bf, wv, bv, Vbf, nullptr, kMKV, kD, kD);

        // attention -> ctx
        attn_kernel<<<dim3(kLQ / 64, kH, kB), 128, 0, stream>>>(Qbf, Kbf, Vbf, ctxbf);

        // x += ctx @ ow^T + ob
        gemm_bf16_kernel<0, 1><<<dim3(kD / 128, kMQ / 128), 256, 0, stream>>>(
            ctxbf, ow, ob, nullptr, x, kMQ, kD, kD);

        // h = LN2(x)  (reuse qin_bf)
        ln_kernel<<<kMQ, 256, 0, stream>>>(x, ln2_g + (size_t)l * kD,
                                           ln2_b + (size_t)l * kD, qin_bf);

        // ff = gelu(h @ w1^T + b1)   (bf16, aliases K/V region)
        gemm_bf16_kernel<1, 0><<<dim3(kDFF / 128, kMQ / 128), 256, 0, stream>>>(
            qin_bf, w1, b1, ffbf, nullptr, kMQ, kDFF, kD);

        // x += ff @ w2^T + b2
        gemm_bf16_kernel<0, 1><<<dim3(kD / 128, kMQ / 128), 256, 0, stream>>>(
            ffbf, w2, b2, nullptr, x, kMQ, kD, kDFF);
    }
}
